// GATDecoder_18167711662122
// MI455X (gfx1250) — matrix operand order repaired and checked by reading, not compiled, not run on hardware
//
#include <hip/hip_runtime.h>

// GAT(1 attention pass) + 3x GraphSAGE + bilinear decoder for MI455X (gfx1250).
// B=8, N=2048, D=8, hidden=16, DEC=128. wave32, WMMA f32_16x16x32_f16.

#define Bm_ 8
#define Nn 2048

typedef __attribute__((ext_vector_type(16))) _Float16 v16h;
typedef __attribute__((ext_vector_type(8)))  _Float16 v8h;
typedef __attribute__((ext_vector_type(8)))  float    v8f;

union V16 { v16h v; v8h h[2]; };

// ---------------------------------------------------------------------------
// Kernel 1: per-node h = x@W, f1 = h@a[:8], f2 = h@a[8:], invdeg = diag(inv_deg)
// ---------------------------------------------------------------------------
__global__ __launch_bounds__(256) void gat_prep(
    const float* __restrict__ x, const float* __restrict__ W,
    const float* __restrict__ av, const float* __restrict__ inv_deg,
    float* __restrict__ f1, float* __restrict__ f2, float* __restrict__ invdeg)
{
  int t = blockIdx.x * 256 + threadIdx.x;   // [0, B*N)
  int b = t >> 11, n = t & (Nn - 1);
  const float* xr = x + (size_t)t * 8;
  float s1 = 0.f, s2 = 0.f;
  #pragma unroll
  for (int d = 0; d < 8; ++d) {
    float h = 0.f;
    #pragma unroll
    for (int k = 0; k < 8; ++k) h += xr[k] * W[k * 8 + d];
    s1 += h * av[d];
    s2 += h * av[8 + d];
  }
  f1[t] = s1;
  f2[t] = s2;
  invdeg[t] = inv_deg[((size_t)b * Nn + n) * Nn + n];
}

// ---------------------------------------------------------------------------
// Kernel 2: softmax row stats. One wave32 per adjacency row j:
//   m[b,j] = max_i e[b,j,i],  rs[b,j] = 1 / sum_i exp(e - m)
// e = adj ? leaky(f1[j]+f2[i], 0.2) : -9e15   (matches reference exactly)
// ---------------------------------------------------------------------------
__global__ __launch_bounds__(256) void gat_stats(
    const int* __restrict__ adj, const float* __restrict__ f1,
    const float* __restrict__ f2, float* __restrict__ mrow,
    float* __restrict__ rsrow)
{
  int wv = threadIdx.x >> 5, lane = threadIdx.x & 31;
  int row = blockIdx.x * 8 + wv;            // [0, B*N)
  int b = row >> 11;
  const int* arow = adj + (size_t)row * Nn;
  const float* f2b = f2 + b * Nn;
  float f1j = f1[row];

  float mx = -3.0e38f;
  for (int i = lane; i < Nn; i += 32) {
    float t = f1j + f2b[i];
    float ev = (arow[i] > 0) ? ((t >= 0.f) ? t : 0.2f * t) : -9.0e15f;
    mx = fmaxf(mx, ev);
  }
  #pragma unroll
  for (int off = 16; off > 0; off >>= 1) mx = fmaxf(mx, __shfl_xor(mx, off, 32));

  float s = 0.f;
  for (int i = lane; i < Nn; i += 32) {
    float t = f1j + f2b[i];
    float ev = (arow[i] > 0) ? ((t >= 0.f) ? t : 0.2f * t) : -9.0e15f;
    s += expf(ev - mx);                     // masked & m finite -> exp(-9e15)=0
  }
  #pragma unroll
  for (int off = 16; off > 0; off >>= 1) s += __shfl_xor(s, off, 32);

  if (lane == 0) { mrow[row] = mx; rsrow[row] = 1.0f / s; }
}

// ---------------------------------------------------------------------------
// Kernel 3: materialize attn^T in f16:  attnT[b][i][j] = attn[b][j][i]
// 64x64 tiles, transpose through LDS; adj is L2-resident from kernel 2.
// ---------------------------------------------------------------------------
__global__ __launch_bounds__(256) void gat_attn_t(
    const int* __restrict__ adj, const float* __restrict__ f1,
    const float* __restrict__ f2, const float* __restrict__ mrow,
    const float* __restrict__ rsrow, _Float16* __restrict__ attnT)
{
  __shared__ _Float16 T[64 * 72];           // [i_local][j_local], padded rows
  int b = blockIdx.z;
  int j0 = blockIdx.y * 64;
  int i0 = blockIdx.x * 64;
  int t = threadIdx.x;
  int c = t & 63, r0 = t >> 6;              // c = i_local for the read phase
  float f2v = f2[b * Nn + i0 + c];

  for (int rr = 0; rr < 16; ++rr) {
    int r = r0 * 16 + rr;                   // j_local
    int j = j0 + r;
    float f1v = f1[b * Nn + j];
    float mv  = mrow[b * Nn + j];
    float rv  = rsrow[b * Nn + j];
    int   am  = adj[((size_t)b * Nn + j) * Nn + i0 + c];
    float tt  = f1v + f2v;
    float ev  = (am > 0) ? ((tt >= 0.f) ? tt : 0.2f * tt) : -9.0e15f;
    T[c * 72 + r] = (_Float16)(expf(ev - mv) * rv);
  }
  __syncthreads();

  int il = t >> 2, q = t & 3;               // 4 threads per output row
  const v8h* src = (const v8h*)(T + il * 72 + q * 16);
  v8h* dst = (v8h*)(attnT + ((size_t)b * Nn + i0 + il) * Nn + j0 + q * 16);
  dst[0] = src[0];
  dst[1] = src[1];
}

// ---------------------------------------------------------------------------
// Kernel 4: SAGE layer via WMMA.
//   newx[i][d] = sum_j attnT[i][j] * X[j][d]   (A = attnT tile, B = X tile)
// Workgroup = 128 thr (4 waves), each wave owns a 16-row i-tile; block = 64 i.
// X^T staged once in LDS (f16, XOR-swizzled 16B chunks -> conflict-free b128).
// Epilogue: *invdeg, @w, self@b, concat, L2-normalize, leaky(0.1).
// ---------------------------------------------------------------------------
template <int DIN>
__global__ __launch_bounds__(128) void sage_layer(
    const float* __restrict__ X,            // (B,N,DIN) f32
    const _Float16* __restrict__ attnT,     // (B,N,N) f16
    const float* __restrict__ invdeg,       // (B,N)
    const float* __restrict__ Wm,           // (DIN,8)
    const float* __restrict__ Bmat,         // (DIN,8)
    float* __restrict__ Y)                  // (B,N,16) f32
{
  __shared__ _Float16 xt[16 * Nn];          // 64KB: swizzled X^T [d][j]
  const int tid = threadIdx.x;
  const int b = blockIdx.y;
  const int i0blk = blockIdx.x * 64;

  // Stage X^T in f16. 16B chunk c of row d stored at chunk (c ^ d).
  for (int idx = tid; idx < 16 * Nn; idx += 128) {
    int j = idx >> 4, d = idx & 15;
    float v = (d < DIN) ? X[((size_t)b * Nn + j) * DIN + d] : 0.0f;
    int sw = ((j >> 3) ^ d) & 255;
    xt[d * Nn + sw * 8 + (j & 7)] = (_Float16)v;
  }
  __syncthreads();

  const int wv = tid >> 5, lane = tid & 31;
  const int i0 = i0blk + wv * 16;
  const int m16 = lane & 15;
  const int hi = lane >> 4;                 // 0 or 1 (lane half)

  v8f acc = {};
  const _Float16* arow = attnT + ((size_t)b * Nn + (i0 + m16)) * Nn;
  const int dcol = m16;                     // B-fragment column (feature d)

  for (int ks = 0; ks < 64; ++ks) {
    int j0 = ks * 32;
    // A frag (16x32 f16): lane<16 -> K = {0..7, 16..23}; lane>=16 -> {8..15, 24..31}
    V16 af;
    af.h[0] = *(const v8h*)(arow + j0 + hi * 8);
    af.h[1] = *(const v8h*)(arow + j0 + 16 + hi * 8);
    // B frag (32x16 f16): lane<16 -> K = {0..7, 16..23}; lane>=16 -> {8..15, 24..31}
    int jb = j0 + hi * 8;
    int c0 = ((jb >> 3) ^ dcol) & 255;
    int c1 = (((jb + 16) >> 3) ^ dcol) & 255;
    V16 bf;
    bf.h[0] = *(const v8h*)(xt + dcol * Nn + c0 * 8);
    bf.h[1] = *(const v8h*)(xt + dcol * Nn + c1 * 8);
    acc = __builtin_amdgcn_wmma_f32_16x16x32_f16(
        false, af.v, false, bf.v, (short)0, acc, false, false);
  }

  __syncthreads();                          // everyone done with xt; reuse LDS
  float* yt = (float*)xt + wv * 16 * 17;    // ytile [i_local][d], pad 17
  #pragma unroll
  for (int r = 0; r < 8; ++r)
    yt[(r + hi * 8) * 17 + m16] = acc[r];
  __syncthreads();

  if (lane < 16) {
    int gi = i0 + lane;
    float sc = invdeg[b * Nn + gi];
    float c[16];
    #pragma unroll
    for (int o = 0; o < 16; ++o) c[o] = 0.f;
    #pragma unroll
    for (int d = 0; d < DIN; ++d) {
      float nv = yt[lane * 17 + d] * sc;
      float xv = X[((size_t)b * Nn + gi) * DIN + d];   // f32 self path
      #pragma unroll
      for (int o = 0; o < 8; ++o) {
        c[o]     += nv * Wm[d * 8 + o];
        c[8 + o] += xv * Bmat[d * 8 + o];
      }
    }
    float n2 = 0.f;
    #pragma unroll
    for (int o = 0; o < 16; ++o) n2 += c[o] * c[o];
    float inv = 1.0f / fmaxf(sqrtf(n2), 1e-12f);
    float* yr = Y + ((size_t)b * Nn + gi) * 16;
    #pragma unroll
    for (int o = 0; o < 16; ++o) {
      float v = c[o] * inv;
      yr[o] = (v >= 0.f) ? v : 0.1f * v;
    }
  }
}

// ---------------------------------------------------------------------------
// Kernel 5: ypred[b] = (P1^T da) . (P2 . (P1^T db))
// ---------------------------------------------------------------------------
__global__ __launch_bounds__(128) void decode_kernel(
    const float* __restrict__ out3, const float* __restrict__ P1,
    const float* __restrict__ P2, float* __restrict__ ypred)
{
  __shared__ float vsh[128];
  __shared__ float psum[128];
  int b = blockIdx.x, e = threadIdx.x;
  const float* da = out3 + ((size_t)b * Nn + (Nn - 2)) * 16;
  const float* db = out3 + ((size_t)b * Nn + (Nn - 1)) * 16;
  float u = 0.f, v = 0.f;
  #pragma unroll
  for (int k = 0; k < 16; ++k) {
    float p = P1[k * 128 + e];
    u += da[k] * p;
    v += db[k] * p;
  }
  vsh[e] = v;
  __syncthreads();
  float w = 0.f;
  for (int f = 0; f < 128; ++f) w += P2[e * 128 + f] * vsh[f];
  psum[e] = u * w;
  __syncthreads();
  for (int s = 64; s > 0; s >>= 1) {
    if (e < s) psum[e] += psum[e + s];
    __syncthreads();
  }
  if (e == 0) ypred[b] = psum[0];
}

// ---------------------------------------------------------------------------
extern "C" void kernel_launch(void* const* d_in, const int* in_sizes, int n_in,
                              void* d_out, int out_size, void* d_ws, size_t ws_size,
                              hipStream_t stream) {
  const float* x       = (const float*)d_in[0];
  const int*   adj     = (const int*)  d_in[1];
  const float* inv_deg = (const float*)d_in[2];
  const float* W       = (const float*)d_in[3];
  const float* av      = (const float*)d_in[4];
  const float* w1      = (const float*)d_in[5];
  const float* b1      = (const float*)d_in[6];
  const float* w2      = (const float*)d_in[7];
  const float* b2      = (const float*)d_in[8];
  const float* w3      = (const float*)d_in[9];
  const float* b3      = (const float*)d_in[10];
  const float* P1      = (const float*)d_in[11];
  const float* P2      = (const float*)d_in[12];

  const size_t BN = (size_t)Bm_ * Nn;            // 16384
  char* ws = (char*)d_ws;
  float*    f1     = (float*)(ws);                               // 64KB
  float*    f2     = (float*)(ws + 1 * BN * 4);
  float*    mrow   = (float*)(ws + 2 * BN * 4);
  float*    rsrow  = (float*)(ws + 3 * BN * 4);
  float*    invdeg = (float*)(ws + 4 * BN * 4);
  float*    buf1   = (float*)(ws + 5 * BN * 4);                  // (B,N,16) f32
  float*    buf2   = (float*)(ws + 5 * BN * 4 + 1 * BN * 64);
  float*    buf3   = (float*)(ws + 5 * BN * 4 + 2 * BN * 64);
  _Float16* attnT  = (_Float16*)(ws + 5 * BN * 4 + 3 * BN * 64); // 67MB f16

  gat_prep <<<BN / 256, 256, 0, stream>>>(x, W, av, inv_deg, f1, f2, invdeg);
  gat_stats<<<BN / 8,   256, 0, stream>>>(adj, f1, f2, mrow, rsrow);
  gat_attn_t<<<dim3(Nn / 64, Nn / 64, Bm_), 256, 0, stream>>>(
      adj, f1, f2, mrow, rsrow, attnT);

  sage_layer<8> <<<dim3(Nn / 64, Bm_), 128, 0, stream>>>(x,    attnT, invdeg, w1, b1, buf1);
  sage_layer<16><<<dim3(Nn / 64, Bm_), 128, 0, stream>>>(buf1, attnT, invdeg, w2, b2, buf2);
  sage_layer<16><<<dim3(Nn / 64, Bm_), 128, 0, stream>>>(buf2, attnT, invdeg, w3, b3, buf3);

  decode_kernel<<<Bm_, 128, 0, stream>>>(buf3, P1, P2, (float*)d_out);
}
